// SpatialAttention_44178033606958
// MI455X (gfx1250) — matrix operand order repaired and checked by reading, not compiled, not run on hardware
//
#include <hip/hip_runtime.h>

typedef __attribute__((ext_vector_type(16))) _Float16 v16h;
typedef __attribute__((ext_vector_type(8)))  _Float16 v8h;
typedef __attribute__((ext_vector_type(8)))  float    v8f;
typedef __attribute__((ext_vector_type(2)))  int      v2i;

#define N_TOK 4096
#define CCH   256
#define HEADS 8
#define HDIM  32
#define NGROUPS 32

#if __has_builtin(__builtin_amdgcn_global_load_async_to_lds_b64) && \
    __has_builtin(__builtin_amdgcn_s_wait_asynccnt)
#define USE_ASYNC_LDS 1
#else
#define USE_ASYNC_LDS 0
#endif

__device__ __forceinline__ v16h cat8(v8h lo, v8h hi) {
  return __builtin_shufflevector(lo, hi, 0,1,2,3,4,5,6,7,8,9,10,11,12,13,14,15);
}
__device__ __forceinline__ v8f wmma_f16(v16h a, v16h b, v8f c) {
  // D = A(16x32 f16) * B(32x16 f16) + C(16x16 f32)
  return __builtin_amdgcn_wmma_f32_16x16x32_f16(false, a, false, b, (short)0, c, false, false);
}

#if USE_ASYNC_LDS
// Builtin prototype (from hipcc diagnostic): (v2i as1*, v2i as3*, imm, imm).
typedef __attribute__((address_space(1))) v2i* as1_v2i;
typedef __attribute__((address_space(3))) v2i* as3_v2i;
// 8-byte async global->LDS copy (per-lane addresses, tracked by ASYNCcnt).
__device__ __forceinline__ void async_cp8(const _Float16* g, _Float16* l) {
  __builtin_amdgcn_global_load_async_to_lds_b64(
      (as1_v2i)(uintptr_t)g,
      (as3_v2i)(uint32_t)(uintptr_t)l,
      0, 0);
}
#endif

// ---------------------------------------------------------------------------
// 1) GroupNorm statistics: one block per (b, group); group = 8 ch x 4096 =
//    32768 contiguous floats.
// ---------------------------------------------------------------------------
__global__ void gn_stats_kernel(const float* __restrict__ x, float* __restrict__ stats) {
  const int bg = blockIdx.x;                       // 64 = B*GROUPS
  const float4* p = (const float4*)(x + (size_t)bg * 32768);
  float s = 0.f, ss = 0.f;
  for (int i = threadIdx.x; i < 8192; i += 256) {
    float4 v = p[i];
    s  += v.x + v.y + v.z + v.w;
    ss += v.x*v.x + v.y*v.y + v.z*v.z + v.w*v.w;
  }
  __shared__ float rs[256], rq[256];
  rs[threadIdx.x] = s; rq[threadIdx.x] = ss;
  __syncthreads();
  for (int off = 128; off > 0; off >>= 1) {
    if (threadIdx.x < off) {
      rs[threadIdx.x] += rs[threadIdx.x + off];
      rq[threadIdx.x] += rq[threadIdx.x + off];
    }
    __syncthreads();
  }
  if (threadIdx.x == 0) {
    float mean = rs[0] * (1.f/32768.f);
    float var  = rq[0] * (1.f/32768.f) - mean*mean;
    stats[bg*2+0] = mean;
    stats[bg*2+1] = rsqrtf(var + 1e-5f);
  }
}

// ---------------------------------------------------------------------------
// 2) GroupNorm apply + transpose: x[B][C][N] fp32 -> xnT[B][N][C] f16
// ---------------------------------------------------------------------------
__global__ void gn_apply_kernel(const float* __restrict__ x, const float* __restrict__ stats,
                                const float* __restrict__ gw, const float* __restrict__ gb,
                                _Float16* __restrict__ xt) {
  const size_t t    = (size_t)blockIdx.x*256 + threadIdx.x;   // 524288 threads
  const size_t base = t * 4;
  const int b   = (int)(base >> 20);            // C*N = 1<<20
  const int rem = (int)(base & 1048575u);
  const int c   = rem >> 12;
  const int n   = rem & 4095;
  const int g   = c >> 3;
  const float mean = stats[(b*NGROUPS + g)*2 + 0];
  const float rstd = stats[(b*NGROUPS + g)*2 + 1];
  const float w    = gw[c] * rstd;
  const float bias = gb[c] - mean * w;
  const float4 v = *(const float4*)(x + base);
  _Float16* dst = xt + ((size_t)b*N_TOK + n)*CCH + c;
  dst[0]     = (_Float16)(v.x*w + bias);
  dst[CCH]   = (_Float16)(v.y*w + bias);
  dst[2*CCH] = (_Float16)(v.z*w + bias);
  dst[3*CCH] = (_Float16)(v.w*w + bias);
}

// ---------------------------------------------------------------------------
// 3) Weight conversion fp32 -> f16; softmax scale baked into q rows of w_qkv.
// ---------------------------------------------------------------------------
__global__ void cvt_w_kernel(const float* __restrict__ wq, const float* __restrict__ wo,
                             _Float16* __restrict__ wq16, _Float16* __restrict__ wo16) {
  const int i = blockIdx.x*256 + threadIdx.x;    // 196608 threads
  float v = wq[i];
  if (i < 65536) v *= 0.17677669529663687f;      // 1/sqrt(32) on q rows
  wq16[i] = (_Float16)v;
  if (i < 65536) wo16[i] = (_Float16)wo[i];
}

// ---------------------------------------------------------------------------
// 4) QKV projection GEMM (M=768, K=256, N=4096 per batch), WMMA f16.
//    One wave = one 16-row o-tile x 4 n-tiles; A cached in registers.
//    All tile/selector math forced scalar via readfirstlane; store path is a
//    single uniform (base, nstride, dstride) descriptor -- no exec masking.
// ---------------------------------------------------------------------------
__global__ void __launch_bounds__(256)
qkv_gemm_kernel(const _Float16* __restrict__ wq16, const _Float16* __restrict__ xt,
                _Float16* __restrict__ Qd, _Float16* __restrict__ Kd,
                _Float16* __restrict__ Vt) {
  const int lane = threadIdx.x & 31;
  const int half = lane >> 4, lm = lane & 15;
  const int task = __builtin_amdgcn_readfirstlane(blockIdx.x*8 + (threadIdx.x >> 5));
  const int b    = task / 3072;                  // 48 o-tiles * 64 n-groups
  const int rem  = task % 3072;
  const int ot   = rem / 64;
  const int ng   = rem % 64;
  const int o0   = ot * 16;

  v16h wa[8];
  const v8h* wrow = (const v8h*)(wq16 + (size_t)(o0 + lm) * CCH);
  #pragma unroll
  for (int ks = 0; ks < 8; ++ks)
    wa[ks] = cat8(wrow[ks*4 + 2*half], wrow[ks*4 + 2*half + 1]);

  const int qkvi = o0 >> 8;                      // 0=q 1=k 2=v (scalar)
  const int h    = (o0 & 255) >> 5;
  const int d0   = (o0 & 31);                    // 0 or 16

  // Uniform store descriptor:
  //   q/k [B][H][N][D]: off = base0 + n*32 + d*1
  //   v   [B][H][D][N]: off = base0 + n*1  + d*4096
  _Float16* basep = (qkvi == 0) ? Qd : (qkvi == 1) ? Kd : Vt;
  const size_t base0   = (size_t)(b*HEADS + h) * (N_TOK*HDIM);
  const size_t nstride = (qkvi == 2) ? (size_t)1     : (size_t)HDIM;
  const size_t dstride = (qkvi == 2) ? (size_t)N_TOK : (size_t)1;

  for (int nt = 0; nt < 4; ++nt) {
    const int n = ng*64 + nt*16 + lm;
    const _Float16* xrow = xt + ((size_t)b*N_TOK + n) * CCH;
    if (nt < 3)
      __builtin_prefetch(xt + ((size_t)b*N_TOK + n + 16) * CCH, 0, 3);

    v16h xb[8];
    #pragma unroll
    for (int ks = 0; ks < 8; ++ks)
      xb[ks] = *(const v16h*)(xrow + ks*32 + half*16);

    v8f acc = {};
    #pragma unroll
    for (int ks = 0; ks < 8; ++ks)
      acc = wmma_f16(wa[ks], xb[ks], acc);

    const size_t rowbase = base0 + (size_t)n*nstride + (size_t)(d0 + half*8)*dstride;
    #pragma unroll
    for (int r = 0; r < 8; ++r)
      basep[rowbase + (size_t)r*dstride] = (_Float16)acc[r];
  }
}

// ---------------------------------------------------------------------------
// 5) Flash attention. Block = 8 waves, one (b,h,128-query chunk).
//    Double-buffered K/Vt tiles in LDS. CDNA5 async global->LDS path
//    (GLOBAL_LOAD_ASYNC_TO_LDS_B64 + ASYNCcnt): fire next tile before the
//    compute phase, wait after; register-staged fallback otherwise. One WG
//    barrier per 32-key iteration. Per-wave P relayout uses a wave-level
//    s_wait_dscnt (same-wave LDS is in-order).
// ---------------------------------------------------------------------------
__global__ void __launch_bounds__(256)
attn_kernel(const _Float16* __restrict__ Q, const _Float16* __restrict__ K,
            const _Float16* __restrict__ Vt, _Float16* __restrict__ O) {
  const int bh     = blockIdx.x >> 5;            // 16 (b*heads)
  const int qchunk = blockIdx.x & 31;            // 32 chunks of 128 queries
  const int wave = __builtin_amdgcn_readfirstlane(threadIdx.x >> 5);
  const int lane = threadIdx.x & 31;
  const int half = lane >> 4, lm = lane & 15;
  const int q0   = qchunk*128 + wave*16;
  const int tid  = threadIdx.x;

  const _Float16* Qb = Q  + (size_t)bh * N_TOK * HDIM;
  const _Float16* Kb = K  + (size_t)bh * N_TOK * HDIM;
  const _Float16* Vb = Vt + (size_t)bh * HDIM * N_TOK;
  _Float16*       Ob = O  + (size_t)bh * N_TOK * HDIM;

  __shared__ __align__(64) _Float16 Kl[2][32*32];     // [buf][key][d]
  __shared__ __align__(64) _Float16 Vl[2][32*32];     // [buf][d][key]
  __shared__ __align__(64) _Float16 Pl[8][16*32];     // per-wave P tile [q][key]

  // Per-thread staging addresses (8B per thread per tile for K and for V).
  const int vrow = tid >> 3, vseg = tid & 7;          // V: row d, 4-half segment

  // Q A-operand: row = q0+lm, two 16B chunks, contiguous k = 16*half + s (same order as K).
  const v8h* qrow = (const v8h*)(Qb + (size_t)(q0 + lm) * HDIM);
  const v16h qa = cat8(qrow[2*half], qrow[2*half + 1]);

  v8f o0v = {}, o1v = {};
  float m[8], l[8];
  #pragma unroll
  for (int r = 0; r < 8; ++r) { m[r] = -3.0e38f; l[r] = 0.f; }

  // Prologue: stage tile 0 into buffer 0.
#if USE_ASYNC_LDS
  async_cp8(Kb + tid*4, Kl[0] + tid*4);
  async_cp8(Vb + (size_t)vrow*N_TOK + vseg*4, Vl[0] + vrow*32 + vseg*4);
  __builtin_amdgcn_s_wait_asynccnt(0);
#else
  ((uint2*)Kl[0])[tid] = ((const uint2*)Kb)[tid];
  *(uint2*)(Vl[0] + vrow*32 + vseg*4) = *(const uint2*)(Vb + (size_t)vrow*N_TOK + vseg*4);
#endif
  __syncthreads();

  for (int it = 0; it < 128; ++it) {
    const int cur = it & 1, nxt = cur ^ 1;
    const int knext = (it + 1) * 32;
    const bool have_next = (it + 1 < 128);

    // Kick off next tile's copy before the compute phase.
#if USE_ASYNC_LDS
    if (have_next) {
      async_cp8(Kb + (size_t)knext*HDIM + tid*4, Kl[nxt] + tid*4);
      async_cp8(Vb + (size_t)vrow*N_TOK + knext + vseg*4, Vl[nxt] + vrow*32 + vseg*4);
    }
#else
    uint2 kreg, vreg;
    if (have_next) {
      kreg = ((const uint2*)(Kb + (size_t)knext * HDIM))[tid];
      vreg = *(const uint2*)(Vb + (size_t)vrow*N_TOK + knext + vseg*4);
    }
#endif

    // S = Q x K^T for 32 keys (two 16-key WMMA tiles).
    const _Float16* Kc = Kl[cur];
    const _Float16* Vc = Vl[cur];
    const v16h kb1 = *(const v16h*)(Kc + lm*32        + half*16);
    const v16h kb2 = *(const v16h*)(Kc + (16 + lm)*32 + half*16);
    v8f z = {};
    v8f s1 = wmma_f16(qa, kb1, z);
    v8f s2 = wmma_f16(qa, kb2, z);

    #pragma unroll
    for (int r = 0; r < 8; ++r) {
      float bm = fmaxf(s1[r], s2[r]);
      #pragma unroll
      for (int off = 1; off < 16; off <<= 1)
        bm = fmaxf(bm, __shfl_xor(bm, off, 16));
      const float mnew  = fmaxf(m[r], bm);
      const float alpha = __expf(m[r] - mnew);
      m[r] = mnew;
      const float p1 = __expf(s1[r] - mnew);
      const float p2 = __expf(s2[r] - mnew);
      float rsum = p1 + p2;
      #pragma unroll
      for (int off = 1; off < 16; off <<= 1)
        rsum += __shfl_xor(rsum, off, 16);
      l[r]   = l[r]*alpha + rsum;
      o0v[r] *= alpha;
      o1v[r] *= alpha;
      // C layout -> [q][key] tile in LDS for A re-layout (wave-private).
      Pl[wave][(half*8 + r)*32 + lm]      = (_Float16)p1;
      Pl[wave][(half*8 + r)*32 + 16 + lm] = (_Float16)p2;
    }

    // Wave-level LDS RAW fence (same-wave DS ops are in-order; no WG barrier).
    asm volatile("s_wait_dscnt 0" ::: "memory");

    const v8h* pr = (const v8h*)(&Pl[wave][lm*32]);
    const v16h pa  = cat8(pr[2*half], pr[2*half + 1]);
    const v16h vb0 = *(const v16h*)(Vc + lm*32        + half*16);
    const v16h vb1 = *(const v16h*)(Vc + (16 + lm)*32 + half*16);
    o0v = wmma_f16(pa, vb0, o0v);   // d = 0..15
    o1v = wmma_f16(pa, vb1, o1v);   // d = 16..31

    // Make the staged tile visible, then barrier.
#if USE_ASYNC_LDS
    if (have_next) __builtin_amdgcn_s_wait_asynccnt(0);
#else
    if (have_next) {
      ((uint2*)Kl[nxt])[tid] = kreg;
      *(uint2*)(Vl[nxt] + vrow*32 + vseg*4) = vreg;
    }
#endif
    __syncthreads();
  }

  #pragma unroll
  for (int r = 0; r < 8; ++r) {
    const float inv = 1.f / l[r];
    const int q = q0 + half*8 + r;
    Ob[(size_t)q*HDIM + lm]      = (_Float16)(o0v[r]*inv);
    Ob[(size_t)q*HDIM + 16 + lm] = (_Float16)(o1v[r]*inv);
  }
}

// ---------------------------------------------------------------------------
// 6) Output projection (M=256, K=256, N=4096 per batch) + residual, fp32 out.
//    Same batched-load + chained-WMMA structure as the QKV GEMM.
// ---------------------------------------------------------------------------
__global__ void __launch_bounds__(256)
out_gemm_kernel(const _Float16* __restrict__ wo16, const _Float16* __restrict__ Oa,
                const float* __restrict__ x, float* __restrict__ out) {
  const int lane = threadIdx.x & 31;
  const int half = lane >> 4, lm = lane & 15;
  const int task = __builtin_amdgcn_readfirstlane(blockIdx.x*8 + (threadIdx.x >> 5));
  const int b    = task / 1024;                  // 16 c-tiles * 64 n-groups
  const int rem  = task % 1024;
  const int ct   = rem / 64;
  const int ng   = rem % 64;
  const int c0   = ct * 16;

  v16h wa[8];
  const v8h* wrow = (const v8h*)(wo16 + (size_t)(c0 + lm) * CCH);
  #pragma unroll
  for (int ks = 0; ks < 8; ++ks)
    wa[ks] = cat8(wrow[ks*4 + 2*half], wrow[ks*4 + 2*half + 1]);

  for (int nt = 0; nt < 4; ++nt) {
    const int n = ng*64 + nt*16 + lm;

    v16h ob[8];
    #pragma unroll
    for (int ks = 0; ks < 8; ++ks) {
      // feature block ks*32..ks*32+31 == head ks; Oa is [B][H][N][D].
      ob[ks] = *(const v16h*)(Oa + (((size_t)b*HEADS + ks)*N_TOK + n)*HDIM + half*16);
      if (nt < 3)
        __builtin_prefetch(Oa + (((size_t)b*HEADS + ks)*N_TOK + n + 16)*HDIM, 0, 3);
    }

    v8f acc = {};
    #pragma unroll
    for (int ks = 0; ks < 8; ++ks)
      acc = wmma_f16(wa[ks], ob[ks], acc);

    #pragma unroll
    for (int r = 0; r < 8; ++r) {
      const int c = c0 + half*8 + r;
      const size_t idx = ((size_t)b*CCH + c)*N_TOK + n;
      out[idx] = x[idx] + acc[r];
    }
  }
}

// ---------------------------------------------------------------------------
extern "C" void kernel_launch(void* const* d_in, const int* in_sizes, int n_in,
                              void* d_out, int out_size, void* d_ws, size_t ws_size,
                              hipStream_t stream) {
  const float* x  = (const float*)d_in[0];   // [2,256,64,64]
  const float* gw = (const float*)d_in[1];   // [256]
  const float* gb = (const float*)d_in[2];   // [256]
  const float* wq = (const float*)d_in[3];   // [768,256]
  const float* wo = (const float*)d_in[4];   // [256,256]
  float* out = (float*)d_out;

  char* ws = (char*)d_ws;
  float*    stats = (float*)(ws + 0);          //    512 B
  _Float16* wq16  = (_Float16*)(ws + 1024);    // 393216 B
  _Float16* wo16  = (_Float16*)(ws + 394240);  // 131072 B
  _Float16* xt    = (_Float16*)(ws + 525312);  // 4 MB  [B][N][C]
  _Float16* Qd    = (_Float16*)(ws + 4719616); // 4 MB  [B][H][N][D]
  _Float16* Kd    = (_Float16*)(ws + 8913920); // 4 MB  [B][H][N][D]
  _Float16* Vt    = (_Float16*)(ws + 13108224);// 4 MB  [B][H][D][N]
  _Float16* Oa    = (_Float16*)(ws + 17302528);// 4 MB  [B][H][N][D]

  gn_stats_kernel<<<64,   256, 0, stream>>>(x, stats);
  gn_apply_kernel<<<2048, 256, 0, stream>>>(x, stats, gw, gb, xt);
  cvt_w_kernel  <<<768,  256, 0, stream>>>(wq, wo, wq16, wo16);
  qkv_gemm_kernel<<<768, 256, 0, stream>>>(wq16, xt, Qd, Kd, Vt);
  attn_kernel   <<<512,  256, 0, stream>>>(Qd, Kd, Vt, Oa);
  out_gemm_kernel<<<256, 256, 0, stream>>>(wo16, Oa, x, out);
}
